// PINN_73615739454025
// MI455X (gfx1250) — matrix operand order repaired and checked by reading, not compiled, not run on hardware
//
#include <hip/hip_runtime.h>
#include <hip/hip_bf16.h>
#include <math.h>

typedef __attribute__((ext_vector_type(16))) _Float16 v16h;
typedef __attribute__((ext_vector_type(8)))  float    v8f;

#define Hd 128          // hidden width
#define Fd 16           // input features
#define INV_RE 1.45e-05f

__device__ __forceinline__ float softplusf(float x) {
    // stable log(1+exp(x)) = max(x,0) + log1p(exp(-|x|))
    return fmaxf(x, 0.0f) + log1pf(expf(-fabsf(x)));
}

// ---------------------------------------------------------------------------
// K0: zero the aggregation buffer (atomics accumulate into it)
// ---------------------------------------------------------------------------
__global__ void k_zero(float* __restrict__ p, long long n) {
    long long i = (long long)blockIdx.x * blockDim.x + threadIdx.x;
    if (i < n) p[i] = 0.0f;
}

// ---------------------------------------------------------------------------
// K1: h = relu(x @ W1 + b1) via v_wmma_f32_16x16x32_f16, error-compensated
//     f16 split (hi/lo) for ~f32 accuracy. One wave = one 16x16 tile of h.
//     Block = 256 threads = 8 waves -> covers all 128 columns of 16 rows.
// ---------------------------------------------------------------------------
__global__ void __launch_bounds__(256) k_h(const float* __restrict__ x,
                                           const float* __restrict__ W1,
                                           const float* __restrict__ b1,
                                           float* __restrict__ h, int n) {
    const int lane     = threadIdx.x & 31;
    const int wave     = threadIdx.x >> 5;     // 0..7 -> column block
    const int halfLane = lane & 15;
    const bool hi      = lane >= 16;
    const int rowBase  = blockIdx.x * 16;
    const int row      = rowBase + halfLane;
    const int col      = wave * 16 + halfLane;

    // --- A: x tile (16 x K), K = 16 real, padded to 32 with zeros.
    // Contiguous k order used here (A and B alike): lanes 0-15 hold K=0..15 (elems 0..15), lanes
    // 16-31 hold K=16..31 (all padding -> zero).
    v16h a_hi = {};
    v16h a_lo = {};
    if (row < n && !hi) {
        const int kBase = 0;
        #pragma unroll
        for (int j = 0; j < 16; ++j) {
            float v  = x[row * Fd + kBase + j];
            _Float16 vh = (_Float16)v;
            a_hi[j] = vh;
            a_lo[j] = (_Float16)(v - (float)vh);
        }
    }

    // --- B: W1 block (K=32 x 16), real K = 16 rows of W1, rest zero.
    // Contiguous k order used here (same as A): lanes 0-15 hold K=0..15 (elems 0..15), lanes
    // 16-31 hold K=16..31 (all padding -> zero).
    v16h b_hi = {};
    v16h b_lo = {};
    if (!hi) {
        #pragma unroll
        for (int j = 0; j < 16; ++j) {
            float v  = W1[j * Hd + col];
            _Float16 vh = (_Float16)v;
            b_hi[j] = vh;
            b_lo[j] = (_Float16)(v - (float)vh);
        }
    }

    // --- C = A*B with hi/lo compensation: hi*hi + hi*lo + lo*hi
    v8f c = {};
    c = __builtin_amdgcn_wmma_f32_16x16x32_f16(false, a_hi, false, b_hi,
                                               (short)0, c, false, false);
    c = __builtin_amdgcn_wmma_f32_16x16x32_f16(false, a_hi, false, b_lo,
                                               (short)0, c, false, false);
    c = __builtin_amdgcn_wmma_f32_16x16x32_f16(false, a_lo, false, b_hi,
                                               (short)0, c, false, false);

    // --- D layout: lanes 0-15 -> N=lane, M=r; lanes 16-31 -> N=lane-16, M=r+8
    const float bias = b1[col];
    #pragma unroll
    for (int r = 0; r < 8; ++r) {
        int m    = hi ? (r + 8) : r;
        int node = rowBase + m;
        if (node < n) {
            float v = c[r] + bias;
            h[(long long)node * Hd + col] = v > 0.0f ? v : 0.0f;
        }
    }
}

// ---------------------------------------------------------------------------
// K2: edge messages + scatter-add.  One "slot" = (edge, 4 columns).
//     agg[dst] += softplus(edge_attr @ We) * h[src]
// ---------------------------------------------------------------------------
__global__ void __launch_bounds__(256) k_edge(const float* __restrict__ h,
                                              const int* __restrict__ src,
                                              const int* __restrict__ dst,
                                              const float* __restrict__ ea,
                                              const float* __restrict__ We,
                                              float* __restrict__ agg, int E) {
    long long gid = (long long)blockIdx.x * blockDim.x + threadIdx.x;
    int e    = (int)(gid >> 5);
    int lane = (int)(gid & 31);
    if (e >= E) return;
    int s = src[e], d = dst[e];
    float a0 = ea[e * 3 + 0], a1 = ea[e * 3 + 1], a2 = ea[e * 3 + 2];
    const long long sBase = (long long)s * Hd;
    const long long dBase = (long long)d * Hd;
    int c0 = lane * 4;
    #pragma unroll
    for (int j = 0; j < 4; ++j) {
        int c = c0 + j;
        float z = a0 * We[c] + a1 * We[Hd + c] + a2 * We[2 * Hd + c];
        float w = softplusf(z);
        atomicAdd(&agg[dBase + c], w * h[sBase + c]);
    }
}

// ---------------------------------------------------------------------------
// K3: h2 = relu(h + agg) (stored in-place into agg); out_sup = h2 @ Wout
// ---------------------------------------------------------------------------
__global__ void __launch_bounds__(256) k_h2(const float* __restrict__ h,
                                            float* __restrict__ agg,
                                            const float* __restrict__ Wout,
                                            float* __restrict__ out_sup, int n) {
    int node = blockIdx.x * blockDim.x + threadIdx.x;
    if (node >= n) return;
    long long base = (long long)node * Hd;
    float o0 = 0.f, o1 = 0.f, o2 = 0.f;
    for (int c = 0; c < Hd; ++c) {
        float v = h[base + c] + agg[base + c];
        v = v > 0.0f ? v : 0.0f;
        agg[base + c] = v;                 // agg becomes h2
        o0 += v * Wout[c * 3 + 0];
        o1 += v * Wout[c * 3 + 1];
        o2 += v * Wout[c * 3 + 2];
    }
    out_sup[node * 3 + 0] = o0;
    out_sup[node * 3 + 1] = o1;
    out_sup[node * 3 + 2] = o2;
}

// ---------------------------------------------------------------------------
// K4: per-point readout with analytic value / gradient / Hessian, then the
//     PDE residuals (domain points) and boundary residuals.
// ---------------------------------------------------------------------------
__global__ void __launch_bounds__(256) k_readout(
        const float* __restrict__ pos, const float* __restrict__ dsp,
        const int* __restrict__ nedges, const float* __restrict__ bsp,
        const int* __restrict__ idxb, const float* __restrict__ h2,
        const float* __restrict__ Wp, const float* __restrict__ Wo2,
        const float* __restrict__ b2, const float* __restrict__ x,
        const float* __restrict__ xm, float* __restrict__ out,
        int Nd_, int Nb_, int oCont, int oMx, int oMy, int oBnd, int oOsd) {
    int p = blockIdx.x * blockDim.x + threadIdx.x;
    int P = Nd_ + Nb_;
    if (p >= P) return;

    float qx, qy;
    int e[3];
    if (p < Nd_) {
        qx = dsp[p * 2 + 0]; qy = dsp[p * 2 + 1];
        e[0] = nedges[p * 3 + 0]; e[1] = nedges[p * 3 + 1]; e[2] = nedges[p * 3 + 2];
    } else {
        int b = p - Nd_;
        qx = bsp[b * 2 + 0]; qy = bsp[b * 2 + 1];
        int nn = idxb[b];
        e[0] = nn; e[1] = nn; e[2] = nn;
    }

    // Per-node geometry: d = max(pos2 - q, 1e-7); subgradient m = 1{diff>1e-7}
    // a(c) = dx*Wp0 + dy*Wp1 + nrm*Wp2
    // da/dx = -mx*Wp0 + gx*Wp2,  gx = -mx*dx/nrm  (same for y)
    // d2a/dx2 = Axx*Wp2 with Axx = mx^2*(1/nrm - dx^2/nrm^3)   (same for yy)
    float dx[3], dy[3], nr[3], Bx[3], By[3], gx[3], gy[3], Axx[3], Ayy[3];
    #pragma unroll
    for (int i = 0; i < 3; ++i) {
        float px = pos[(long long)e[i] * 3 + 0];
        float py = pos[(long long)e[i] * 3 + 1];
        float rx = px - qx, ry = py - qy;
        float mx = rx > 1e-7f ? 1.0f : 0.0f;
        float my = ry > 1e-7f ? 1.0f : 0.0f;
        float ddx = fmaxf(rx, 1e-7f), ddy = fmaxf(ry, 1e-7f);
        float n2  = ddx * ddx + ddy * ddy;
        float n   = sqrtf(n2);
        float inv = 1.0f / n;
        float inv3 = inv * inv * inv;
        dx[i] = ddx; dy[i] = ddy; nr[i] = n;
        Bx[i] = -mx; By[i] = -my;
        gx[i] = -mx * ddx * inv;
        gy[i] = -my * ddy * inv;
        Axx[i] = mx * mx * (inv - ddx * ddx * inv3);
        Ayy[i] = my * my * (inv - ddy * ddy * inv3);
    }

    float o[3]  = {0, 0, 0}, Gx[3] = {0, 0, 0}, Gy[3] = {0, 0, 0};
    float Hxx[3] = {0, 0, 0}, Hyy[3] = {0, 0, 0};

    for (int c = 0; c < Hd; ++c) {
        float w0 = Wp[c], w1 = Wp[Hd + c], w2 = Wp[2 * Hd + c];
        float feat = 0, Fx = 0, Fy = 0, Fxx = 0, Fyy = 0;
        #pragma unroll
        for (int i = 0; i < 3; ++i) {
            float a    = dx[i] * w0 + dy[i] * w1 + nr[i] * w2;
            float sig  = 1.0f / (1.0f + expf(-a));
            float sp   = softplusf(a);
            float dax  = Bx[i] * w0 + gx[i] * w2;
            float day  = By[i] * w1 + gy[i] * w2;
            float sigp = sig * (1.0f - sig);
            float hv   = h2[(long long)e[i] * Hd + c];
            feat += sp * hv;
            Fx   += sig * dax * hv;
            Fy   += sig * day * hv;
            Fxx  += (sigp * dax * dax + sig * Axx[i] * w2) * hv;
            Fyy  += (sigp * day * day + sig * Ayy[i] * w2) * hv;
        }
        float t   = tanhf(feat);
        float u1  = 1.0f - t * t;
        float tx  = u1 * Fx, ty = u1 * Fy;
        float txx = u1 * Fxx - 2.0f * t * u1 * Fx * Fx;
        float tyy = u1 * Fyy - 2.0f * t * u1 * Fy * Fy;
        #pragma unroll
        for (int oo = 0; oo < 3; ++oo) {
            float wv = Wo2[c * 3 + oo];
            o[oo]   += t   * wv;
            Gx[oo]  += tx  * wv;
            Gy[oo]  += ty  * wv;
            Hxx[oo] += txx * wv;
            Hyy[oo] += tyy * wv;
        }
    }
    o[0] += b2[0]; o[1] += b2[1]; o[2] += b2[2];

    if (p < Nd_) {
        out[oOsd + p * 3 + 0] = o[0];
        out[oOsd + p * 3 + 1] = o[1];
        out[oOsd + p * 3 + 2] = o[2];
        out[oCont + p] = Gx[0] + Gy[1];                          // u_x + v_y
        out[oMx + p] = o[0] * Gx[0] + o[1] * Gy[0] + Gx[2]
                       - (Hxx[0] + Hyy[0]) * INV_RE;
        out[oMy + p] = o[0] * Gx[1] + o[1] * Gy[1] + Gy[2]
                       - (Hxx[1] + Hyy[1]) * INV_RE;
    } else {
        int b = p - Nd_;
        int node = e[0];
        const float* xb = x + (long long)node * Fd;
        const float* mb = xm + (long long)node * 6;
        float tx_ = xb[0], ty_ = xb[1];
        float nx = -ty_, ny = tx_;
        float ub = o[0], vb = o[1], pb = o[2];
        float ub_x = Gx[0], vb_y = Gy[1], pb_x = Gx[2], pb_y = Gy[2];
        float r_vt   = fabsf(tx_ * ub + ty_ * vb - xb[2]) * mb[1];
        float r_vn   = fabsf(nx * ub + ny * vb - xb[3]) * mb[2];
        float r_p    = fabsf(pb - xb[4]) * mb[3];
        float r_dvdn = fabsf(nx * ub_x + ny * vb_y - xb[5]) * mb[4];
        float r_dpdn = fabsf(nx * pb_x + ny * pb_y - xb[6]) * mb[5];
        float cden   = mb[1] + mb[2] + mb[3] + mb[4] + mb[5];
        out[oBnd + b] = (r_vt + r_vn + r_p + r_dvdn + r_dpdn) / cden;
    }
}

// ---------------------------------------------------------------------------
extern "C" void kernel_launch(void* const* d_in, const int* in_sizes, int n_in,
                              void* d_out, int out_size, void* d_ws, size_t ws_size,
                              hipStream_t stream) {
    const float* x     = (const float*)d_in[0];
    const float* xmask = (const float*)d_in[1];
    const int*   eidx  = (const int*)  d_in[2];
    const float* eattr = (const float*)d_in[3];
    const float* pos   = (const float*)d_in[4];
    // d_in[5] = batch (unused)
    const float* dsp   = (const float*)d_in[6];
    const int*   nedg  = (const int*)  d_in[7];
    const float* bsp   = (const float*)d_in[8];
    const int*   idxb  = (const int*)  d_in[9];
    const float* W1    = (const float*)d_in[10];
    const float* b1    = (const float*)d_in[11];
    const float* We    = (const float*)d_in[12];
    const float* Wout  = (const float*)d_in[13];
    const float* Wp    = (const float*)d_in[14];
    const float* Wo2   = (const float*)d_in[15];
    const float* b2    = (const float*)d_in[16];

    const int N_  = in_sizes[0] / Fd;       // 100000
    const int E_  = in_sizes[3] / 3;        // 800000
    const int Nd_ = in_sizes[6] / 2;        // 16384
    const int Nb_ = in_sizes[9];            // 4096

    // Workspace: h (N*128 f32) | agg/h2 (N*128 f32)  -> ~102.4 MB total
    float* h   = (float*)d_ws;
    float* agg = h + (long long)N_ * Hd;
    float* out = (float*)d_out;

    // Output layout (tuple order, flattened):
    const int oSup  = 0;                 // (N,3)
    const int oCont = N_ * 3;            // (Nd,)
    const int oMx   = oCont + Nd_;       // (Nd,)
    const int oMy   = oMx + Nd_;         // (Nd,)
    const int oBnd  = oMy + Nd_;         // (Nb,)
    const int oOsd  = oBnd + Nb_;        // (Nd,3)

    const long long aggN = (long long)N_ * Hd;
    k_zero<<<(int)((aggN + 255) / 256), 256, 0, stream>>>(agg, aggN);

    k_h<<<(N_ + 15) / 16, 256, 0, stream>>>(x, W1, b1, h, N_);

    const long long slots = (long long)E_ * 32;
    k_edge<<<(int)((slots + 255) / 256), 256, 0, stream>>>(
        h, eidx, eidx + E_, eattr, We, agg, E_);

    k_h2<<<(N_ + 255) / 256, 256, 0, stream>>>(h, agg, Wout, out + oSup, N_);

    const int P = Nd_ + Nb_;
    k_readout<<<(P + 255) / 256, 256, 0, stream>>>(
        pos, dsp, nedg, bsp, idxb, agg /* = h2 */, Wp, Wo2, b2, x, xmask,
        out, Nd_, Nb_, oCont, oMx, oMy, oBnd, oOsd);
}
